// EncoderLayer_61383672594905
// MI455X (gfx1250) — compile-verified, matrix operand order checked by reading, not run on hardware
//
#include <hip/hip_runtime.h>
#include <hip/hip_bf16.h>
#include <math.h>
#include <stdint.h>

typedef __attribute__((ext_vector_type(16))) _Float16 v16h;
typedef __attribute__((ext_vector_type(8)))  _Float16 v8h;
typedef __attribute__((ext_vector_type(8)))  float    v8f;
typedef unsigned int u32x4 __attribute__((ext_vector_type(4)));
typedef int          i32x8 __attribute__((ext_vector_type(8)));
typedef int          i32x4 __attribute__((ext_vector_type(4)));

union V16U { v16h v; v8h h[2]; };

#if __has_builtin(__builtin_amdgcn_tensor_load_to_lds) && \
    __has_builtin(__builtin_amdgcn_s_wait_tensorcnt)
#define HAVE_TDM 1
#endif

__device__ __forceinline__ v8f wmma16(v16h a, v16h b, v8f c) {
  return __builtin_amdgcn_wmma_f32_16x16x32_f16(false, a, false, b, (short)0, c,
                                                false, false);
}

// 16x32 f16 fragment load (CDNA5 ISA 7.12.2). Works for A and for B when B is
// staged transposed (LDS row = output column, contiguous = K).
__device__ __forceinline__ v16h ldfrag(const _Float16* base, int stride, int k0) {
  int lane = threadIdx.x & 31;
  int r    = lane & 15;
  int kb   = k0 + ((lane >> 4) << 3);
  V16U u;
  u.h[0] = *(const v8h*)(base + r * stride + kb);
  u.h[1] = *(const v8h*)(base + r * stride + kb + 16);
  return u.v;
}

__device__ __forceinline__ float mishf(float x) {
  float sp = (x > 20.f) ? x : log1pf(expf(x));
  return x * tanhf(sp);
}

#ifdef HAVE_TDM
// Issue one 2D TDM tile load: tile_d1 rows x tile_d0 f16 elements, global row
// stride `stride0` elems. pad_int/pad_amt (D# codes) make the TDM append
// LDS padding each row: interval 2^(pad_int+1) DWORDs, amount (pad_amt+1) DWORDs.
__device__ __forceinline__ void tdm_load_2d(const void* lds_dst,
                                            const _Float16* gsrc,
                                            unsigned tile_d0, unsigned tile_d1,
                                            unsigned stride0,
                                            unsigned pad_int, unsigned pad_amt) {
  unsigned long long ga = (unsigned long long)(uintptr_t)gsrc;
  u32x4 g0;
  g0[0] = 1u;                                  // D# count=1, user mode
  g0[1] = (unsigned)(uintptr_t)lds_dst;        // lds_addr (bytes, LDS offset)
  g0[2] = (unsigned)ga;                        // global_addr[31:0]
  g0[3] = (unsigned)((ga >> 32) & 0x01FFFFFFull) | (2u << 30); // [56:32]|type=2
  i32x8 g1;
  g1[0] = (int)((1u << 16) | (1u << 20) | (pad_int << 22) | (pad_amt << 25));
  //            data_size=2B   pad_enable
  g1[1] = (int)((tile_d0 & 0xFFFFu) << 16);    // tensor_dim0 = tile_d0
  g1[2] = (int)((tile_d1 & 0xFFFFu) << 16);    // tensor_dim1 = tile_d1
  g1[3] = (int)((tile_d0 & 0xFFFFu) << 16);    // tile_dim0
  g1[4] = (int)(tile_d1 & 0xFFFFu);            // tile_dim1 (tile_dim2 = 0 -> 2D)
  g1[5] = (int)stride0;                        // tensor_dim0_stride[31:0]
  g1[6] = 0;
  g1[7] = 0;
  i32x4 z4 = {0, 0, 0, 0};
#if __clang_major__ >= 23
  i32x8 z8 = {0, 0, 0, 0, 0, 0, 0, 0};
  __builtin_amdgcn_tensor_load_to_lds(g0, g1, z4, z4, z8, 0);
#else
  __builtin_amdgcn_tensor_load_to_lds(g0, g1, z4, z4, 0);
#endif
}
#else
__device__ __forceinline__ void stage_fb(_Float16* lds, int lstride,
                                         const _Float16* g, int gstride,
                                         int d0, int d1) {
  int cpr = d0 >> 3;
  int total = cpr * d1;
  for (int c = threadIdx.x; c < total; c += 256) {
    int row = c / cpr;
    int off = (c - row * cpr) << 3;
    *(v8h*)(lds + row * lstride + off) =
        *(const v8h*)(g + (size_t)row * gstride + off);
  }
}
#endif

// Issue (no wait): stage a d1 x d0 f16 tile into LDS, row stride d0+8 halves.
__device__ __forceinline__ void issue_tile(_Float16* lds, int lstride,
                                           const _Float16* g, int gstride,
                                           int d0, int d1, unsigned pad_int,
                                           int wave) {
#ifdef HAVE_TDM
  (void)lstride;
  if (wave == 0)
    tdm_load_2d(lds, g, (unsigned)d0, (unsigned)d1, (unsigned)gstride, pad_int,
                3u);
#else
  (void)pad_int; (void)wave;
  stage_fb(lds, lstride, g, gstride, d0, d1);
#endif
}

__device__ __forceinline__ void tile_wait(int wave) {
#ifdef HAVE_TDM
  if (wave == 0) __builtin_amdgcn_s_wait_tensorcnt(0);
#else
  (void)wave;
#endif
}

// ---------------------------------------------------------------------------
// prep kernels
// ---------------------------------------------------------------------------
__global__ __launch_bounds__(256) void cvt_kernel(const float* __restrict__ in,
                                                  _Float16* __restrict__ out) {
  int i = (blockIdx.x * 256 + threadIdx.x) * 4;
  float4 f = *(const float4*)(in + i);
  out[i + 0] = (_Float16)f.x;
  out[i + 1] = (_Float16)f.y;
  out[i + 2] = (_Float16)f.z;
  out[i + 3] = (_Float16)f.w;
}

// Wt[n][kk] = (f16) W[kk][n]
__global__ __launch_bounds__(256) void wprep_kernel(const float* __restrict__ W,
                                                    _Float16* __restrict__ Wt,
                                                    int K, int N) {
  int i = blockIdx.x * 256 + threadIdx.x;  // grid covers K*N exactly
  int kk = i / N, n = i - kk * N;
  Wt[(size_t)n * K + kk] = (_Float16)W[i];
}

// ---------------------------------------------------------------------------
// GEMM: out = act(A[M,K](f16) @ Bt[N,K]^T(f16) + bias) * oscale
// Cf: optional f32 out; Ch: optional f16 out (or V^T layout when vt=1).
// Block tile 128x128, 8 waves (4x2), wave tile 32x64, K-step 64.
// TDM double-buffered: issue tiles for stage i+1, compute stage i, wait at swap.
// ---------------------------------------------------------------------------
#define GEMM_LDS 73728  // 2 x [128][72] f16 for A and B

__global__ __launch_bounds__(256) void gemm_kernel(
    const _Float16* __restrict__ A, const _Float16* __restrict__ Bt,
    const float* __restrict__ bias, float* __restrict__ Cf,
    _Float16* __restrict__ Ch,
    int M, int K, int N, int act, float oscale, int vt) {
  extern __shared__ char gsm[];
  _Float16* As = (_Float16*)gsm;            // [2][128][72]
  _Float16* Bs = (_Float16*)(gsm + 36864);  // [2][128][72]
  int tid = threadIdx.x;
  int lane = tid & 31, w = tid >> 5;
  int wm = w & 3, wn = w >> 2;
  int m0 = blockIdx.y * 128, n0 = blockIdx.x * 128;

  v8f acc[2][4];
  const v8f vzero = {0.f, 0.f, 0.f, 0.f, 0.f, 0.f, 0.f, 0.f};
#pragma unroll
  for (int i = 0; i < 2; ++i)
#pragma unroll
    for (int j = 0; j < 4; ++j) acc[i][j] = vzero;

  // prologue: stage k-tile 0
  issue_tile(As, 72, A + (size_t)m0 * K, K, 64, 128, 4u, w);
  issue_tile(Bs, 72, Bt + (size_t)n0 * K, K, 64, 128, 4u, w);
  tile_wait(w);
  __syncthreads();

  int nk = K >> 6;
  for (int ki = 0; ki < nk; ++ki) {
    int cur = ki & 1;
    const _Float16* Ac = As + cur * 9216;
    const _Float16* Bc = Bs + cur * 9216;
    if (ki + 1 < nk) {  // prefetch next k-tile into the other buffer
      int k1 = (ki + 1) << 6;
      issue_tile(As + (1 - cur) * 9216, 72, A + (size_t)m0 * K + k1, K, 64, 128,
                 4u, w);
      issue_tile(Bs + (1 - cur) * 9216, 72, Bt + (size_t)n0 * K + k1, K, 64,
                 128, 4u, w);
    }
#pragma unroll
    for (int ks = 0; ks < 64; ks += 32) {
      v16h af[2], bf[4];
#pragma unroll
      for (int mt = 0; mt < 2; ++mt)
        af[mt] = ldfrag(Ac + (wm * 32 + mt * 16) * 72, 72, ks);
#pragma unroll
      for (int nt = 0; nt < 4; ++nt)
        bf[nt] = ldfrag(Bc + (wn * 64 + nt * 16) * 72, 72, ks);
#pragma unroll
      for (int mt = 0; mt < 2; ++mt)
#pragma unroll
        for (int nt = 0; nt < 4; ++nt)
          acc[mt][nt] = wmma16(af[mt], bf[nt], acc[mt][nt]);
    }
    __syncthreads();   // all waves done reading buffer `cur`
    tile_wait(w);      // next buffer's DMA complete
    __syncthreads();   // release
  }

  int mloc = (lane >> 4) << 3, nloc = lane & 15;
#pragma unroll
  for (int mt = 0; mt < 2; ++mt)
#pragma unroll
    for (int nt = 0; nt < 4; ++nt) {
      int mb = m0 + wm * 32 + mt * 16 + mloc;
      int nb = n0 + wn * 64 + nt * 16 + nloc;
#pragma unroll
      for (int r = 0; r < 8; ++r) {
        float val = acc[mt][nt][r] + bias[nb];
        if (act) val = mishf(val);
        val *= oscale;
        if (vt) {
          // write V^T[b][h][n][s] for attention B-operand
          int tok = mb + r, bb = tok >> 10, s = tok & 1023;
          int hh = nb >> 6, nn = nb & 63;
          Ch[(((size_t)(bb * 8 + hh)) * 64 + nn) * 1024 + s] = (_Float16)val;
        } else {
          if (Cf) Cf[(size_t)(mb + r) * N + nb] = val;
          if (Ch) Ch[(size_t)(mb + r) * N + nb] = (_Float16)val;
        }
      }
    }
}

// ---------------------------------------------------------------------------
// Fused attention: block = (b, h, 16-query tile).
// scores (WMMA) -> entmax1.5 (bisection for tau) -> attn @ V (WMMA).
// q is pre-scaled by 1/16 (1/sqrt(hd) * entmax z/2) at projection time.
// K/V chunks are TDM double-buffered; first V DMA overlaps entmax.
// ---------------------------------------------------------------------------
#define ATTN_LDS 141824

__global__ __launch_bounds__(256) void attn_kernel(
    const _Float16* __restrict__ qh, const _Float16* __restrict__ kh,
    const _Float16* __restrict__ vT, _Float16* __restrict__ ctxh) {
  const int S = 1024, D = 512, HD = 64, H = 8;
  extern __shared__ char smem[];
  float*    Z   = (float*)(smem);               // [16][1024] f32
  _Float16* P   = (_Float16*)(smem + 65536);    // [16][1032] f16
  _Float16* stg = (_Float16*)(smem + 98560);    // [2][9216] halves (K/V chunks)
  _Float16* qld = (_Float16*)(smem + 135424);   // [16][72]
  float*    red = (float*)(smem + 137728);      // [4][8][32]

  int tid = threadIdx.x, lane = tid & 31, w = tid >> 5;
  int blk = blockIdx.x;
  int qb = blk & 63, h = (blk >> 6) & 7, b = blk >> 9;
  int q0 = qb << 4;
  size_t tok0 = (size_t)b * S;
  const v8f vzero = {0.f, 0.f, 0.f, 0.f, 0.f, 0.f, 0.f, 0.f};

  // prologue: Q tile + first K chunk
  issue_tile(qld, 72, qh + (tok0 + q0) * D + h * HD, D, 64, 16, 4u, w);
  issue_tile(stg, 72, kh + tok0 * D + h * HD, D, 64, 128, 4u, w);
  tile_wait(w);
  __syncthreads();

  // ---- z = q_scaled . k^T into Z[16][1024], pipelined over 8 key chunks
  for (int ci = 0; ci < 8; ++ci) {
    int cur = ci & 1;
    const _Float16* sc = stg + cur * 9216;
    if (ci + 1 < 8)
      issue_tile(stg + (1 - cur) * 9216, 72,
                 kh + (tok0 + (ci + 1) * 128) * D + h * HD, D, 64, 128, 4u, w);
    v8f acc = vzero;
#pragma unroll
    for (int kk0 = 0; kk0 < 64; kk0 += 32) {
      v16h a  = ldfrag(qld, 72, kk0);
      v16h bf = ldfrag(sc + (w * 16) * 72, 72, kk0);
      acc = wmma16(a, bf, acc);
    }
    int mloc = (lane >> 4) << 3, nloc = lane & 15;
#pragma unroll
    for (int r = 0; r < 8; ++r)
      Z[(mloc + r) * 1024 + ci * 128 + w * 16 + nloc] = acc[r];
    __syncthreads();
    tile_wait(w);
    __syncthreads();
  }

  // first V chunk streams in underneath the entmax bisection
  const _Float16* vbase = vT + ((size_t)(b * H + h) * HD) * S;
  issue_tile(stg, 136, vbase, S, 128, 64, 5u, w);

  // ---- entmax1.5: 16 lanes/row; tau s.t. sum relu(z - tau)^2 = 1 (monotone)
  {
    int row = tid >> 4, sl = tid & 15;
    float zr[64];
    float m = -3.0e38f;
#pragma unroll
    for (int i = 0; i < 64; ++i) {
      float t = Z[row * 1024 + sl + i * 16];
      zr[i] = t;
      m = fmaxf(m, t);
    }
#pragma unroll
    for (int off = 8; off > 0; off >>= 1) m = fmaxf(m, __shfl_xor(m, off, 32));
#pragma unroll
    for (int i = 0; i < 64; ++i) zr[i] -= m;
    float lo = -1.0f, hi = 0.0f;
    for (int it = 0; it < 32; ++it) {
      float tm = 0.5f * (lo + hi);
      float s = 0.f;
#pragma unroll
      for (int i = 0; i < 64; ++i) {
        float d = fmaxf(zr[i] - tm, 0.f);
        s = fmaf(d, d, s);
      }
#pragma unroll
      for (int off = 8; off > 0; off >>= 1) s += __shfl_xor(s, off, 32);
      if (s >= 1.0f) lo = tm; else hi = tm;
    }
    float tau = 0.5f * (lo + hi);
#pragma unroll
    for (int i = 0; i < 64; ++i) {
      float d = fmaxf(zr[i] - tau, 0.f);
      P[row * 1032 + sl + i * 16] = (_Float16)(d * d);
    }
  }
  __syncthreads();
  tile_wait(w);
  __syncthreads();

  // ---- ctx = P(16x1024) @ V_head(1024x64); 4 n-tiles x 2 k-halves, pipelined
  int ntile = w & 3, khalf = w >> 2;
  v8f vacc = vzero;
  for (int ci = 0; ci < 8; ++ci) {
    int cur = ci & 1;
    const _Float16* sv = stg + cur * 9216;
    if (ci + 1 < 8)
      issue_tile(stg + (1 - cur) * 9216, 136, vbase + (ci + 1) * 128, S, 128,
                 64, 5u, w);
#pragma unroll
    for (int sub = 0; sub < 64; sub += 32) {
      int kl = khalf * 64 + sub;
      v16h a  = ldfrag(P, 1032, ci * 128 + kl);
      v16h bf = ldfrag(sv + (ntile * 16) * 136, 136, kl);
      vacc = wmma16(a, bf, vacc);
    }
    __syncthreads();
    tile_wait(w);
    __syncthreads();
  }
  if (khalf == 1) {
#pragma unroll
    for (int r = 0; r < 8; ++r) red[(ntile * 8 + r) * 32 + lane] = vacc[r];
  }
  __syncthreads();
  if (khalf == 0) {
    int mloc = (lane >> 4) << 3, nloc = lane & 15;
#pragma unroll
    for (int r = 0; r < 8; ++r) {
      float val = vacc[r] + red[(ntile * 8 + r) * 32 + lane];
      ctxh[(tok0 + q0 + mloc + r) * D + h * HD + ntile * 16 + nloc] =
          (_Float16)val;
    }
  }
}

// ---------------------------------------------------------------------------
// out = xres + LayerNorm(y)*g + b (f32), optional f16 copy for next GEMM.
// ---------------------------------------------------------------------------
__global__ __launch_bounds__(256) void ln_res_kernel(
    const float* __restrict__ xres, const float* __restrict__ y,
    const float* __restrict__ g, const float* __restrict__ be,
    float* __restrict__ out, _Float16* __restrict__ outh) {
  const int D = 512;
  __shared__ float ps[8], pss[8];
  int row = blockIdx.x, tid = threadIdx.x;
  size_t base = (size_t)row * D;
  float v0 = y[base + tid], v1 = y[base + tid + 256];
  float s = v0 + v1, ss = v0 * v0 + v1 * v1;
#pragma unroll
  for (int off = 16; off > 0; off >>= 1) {
    s += __shfl_xor(s, off, 32);
    ss += __shfl_xor(ss, off, 32);
  }
  if ((tid & 31) == 0) { ps[tid >> 5] = s; pss[tid >> 5] = ss; }
  __syncthreads();
  if (tid == 0) {
    float S = 0.f, SS = 0.f;
    for (int i = 0; i < 8; ++i) { S += ps[i]; SS += pss[i]; }
    float mu = S * (1.f / 512.f);
    float var = SS * (1.f / 512.f) - mu * mu;
    ps[0] = mu;
    pss[0] = rsqrtf(var + 1e-5f);
  }
  __syncthreads();
  float mu = ps[0], rs = pss[0];
  float o0 = xres[base + tid]       + (v0 - mu) * rs * g[tid]       + be[tid];
  float o1 = xres[base + tid + 256] + (v1 - mu) * rs * g[tid + 256] + be[tid + 256];
  out[base + tid] = o0;
  out[base + tid + 256] = o1;
  if (outh) {
    outh[base + tid] = (_Float16)o0;
    outh[base + tid + 256] = (_Float16)o1;
  }
}

// ---------------------------------------------------------------------------
extern "C" void kernel_launch(void* const* d_in, const int* in_sizes, int n_in,
                              void* d_out, int out_size, void* d_ws,
                              size_t ws_size, hipStream_t stream) {
  (void)in_sizes; (void)n_in; (void)out_size; (void)ws_size;
  const float* x   = (const float*)d_in[0];
  const float* Wq  = (const float*)d_in[1];
  const float* bq  = (const float*)d_in[2];
  const float* Wk  = (const float*)d_in[3];
  const float* bk  = (const float*)d_in[4];
  const float* Wv  = (const float*)d_in[5];
  const float* bv  = (const float*)d_in[6];
  const float* Wo  = (const float*)d_in[7];
  const float* bo  = (const float*)d_in[8];
  const float* g1  = (const float*)d_in[9];
  const float* be1 = (const float*)d_in[10];
  const float* W1  = (const float*)d_in[11];
  const float* b1  = (const float*)d_in[12];
  const float* W2  = (const float*)d_in[13];
  const float* b2  = (const float*)d_in[14];
  const float* g2  = (const float*)d_in[15];
  const float* be2 = (const float*)d_in[16];
  float* out = (float*)d_out;

  char* wsb = (char*)d_ws;
  _Float16* xh   = (_Float16*)(wsb + 0);         // 8 MiB
  _Float16* WqT  = (_Float16*)(wsb + 8388608);
  _Float16* WkT  = (_Float16*)(wsb + 8912896);
  _Float16* WvT  = (_Float16*)(wsb + 9437184);
  _Float16* WoT  = (_Float16*)(wsb + 9961472);
  _Float16* W1T  = (_Float16*)(wsb + 10485760);  // 2 MiB
  _Float16* W2T  = (_Float16*)(wsb + 12582912);  // 2 MiB
  _Float16* qh   = (_Float16*)(wsb + 14680064);  // 8 MiB, overlaid by ffh
  _Float16* kh   = (_Float16*)(wsb + 23068672);  // 8 MiB, overlaid by ffh
  _Float16* vT   = (_Float16*)(wsb + 31457280);  // 8 MiB, overlaid by ffh
  _Float16* ctxh = (_Float16*)(wsb + 39845888);  // 8 MiB, overlaid by ffh
  _Float16* ffh  = (_Float16*)(wsb + 14680064);  // 32 MiB overlay
  float*    yb   = (float*)(wsb + 48234496);     // 16 MiB
  float*    x1   = (float*)(wsb + 65011712);     // 16 MiB
  _Float16* x1h  = (_Float16*)(wsb + 81788928);  // 8 MiB

  (void)hipFuncSetAttribute(reinterpret_cast<const void*>(&attn_kernel),
                            hipFuncAttributeMaxDynamicSharedMemorySize,
                            ATTN_LDS);
  (void)hipFuncSetAttribute(reinterpret_cast<const void*>(&gemm_kernel),
                            hipFuncAttributeMaxDynamicSharedMemorySize,
                            GEMM_LDS);

  dim3 blk(256);
  // one-time f16 prep
  cvt_kernel<<<dim3(4096), blk, 0, stream>>>(x, xh);
  wprep_kernel<<<dim3(1024), blk, 0, stream>>>(Wq, WqT, 512, 512);
  wprep_kernel<<<dim3(1024), blk, 0, stream>>>(Wk, WkT, 512, 512);
  wprep_kernel<<<dim3(1024), blk, 0, stream>>>(Wv, WvT, 512, 512);
  wprep_kernel<<<dim3(1024), blk, 0, stream>>>(Wo, WoT, 512, 512);
  wprep_kernel<<<dim3(4096), blk, 0, stream>>>(W1, W1T, 512, 2048);
  wprep_kernel<<<dim3(4096), blk, 0, stream>>>(W2, W2T, 2048, 512);
  // QKV projections (q pre-scaled by 1/16; v written transposed per head)
  gemm_kernel<<<dim3(4, 64), blk, GEMM_LDS, stream>>>(
      xh, WqT, bq, nullptr, qh, 8192, 512, 512, 0, 0.0625f, 0);
  gemm_kernel<<<dim3(4, 64), blk, GEMM_LDS, stream>>>(
      xh, WkT, bk, nullptr, kh, 8192, 512, 512, 0, 1.0f, 0);
  gemm_kernel<<<dim3(4, 64), blk, GEMM_LDS, stream>>>(
      xh, WvT, bv, nullptr, vT, 8192, 512, 512, 0, 1.0f, 1);
  // fused entmax attention
  attn_kernel<<<dim3(4096), blk, ATTN_LDS, stream>>>(qh, kh, vT, ctxh);
  // output projection + residual LN
  gemm_kernel<<<dim3(4, 64), blk, GEMM_LDS, stream>>>(
      ctxh, WoT, bo, yb, nullptr, 8192, 512, 512, 0, 1.0f, 0);
  ln_res_kernel<<<dim3(8192), blk, 0, stream>>>(x, yb, g1, be1, x1, x1h);
  // FFN (Mish) + residual LN
  gemm_kernel<<<dim3(16, 64), blk, GEMM_LDS, stream>>>(
      x1h, W1T, b1, nullptr, ffh, 8192, 512, 2048, 1, 1.0f, 0);
  gemm_kernel<<<dim3(4, 64), blk, GEMM_LDS, stream>>>(
      ffh, W2T, b2, yb, nullptr, 8192, 2048, 512, 0, 1.0f, 0);
  ln_res_kernel<<<dim3(8192), blk, 0, stream>>>(x1, yb, g2, be2, out, nullptr);
}
